// BA_Predict_8297876816437
// MI455X (gfx1250) — matrix operand order repaired and checked by reading, not compiled, not run on hardware
//
#include <hip/hip_runtime.h>
#include <hip/hip_bf16.h>
#include <stdint.h>

// ---------------------------------------------------------------------------
// MI455X (gfx1250, wave32). Dense math -> V_WMMA_F32_16X16X32_BF16 GEMM over
// pre-packed bf16 operands; tile staging via the Tensor Data Mover
// (tensor_load_to_lds + s_wait_tensorcnt) with a vectorized fallback.
// ---------------------------------------------------------------------------

typedef __attribute__((ext_vector_type(16))) __bf16 v16bf;
typedef __attribute__((ext_vector_type(8)))  __bf16 v8bf;
typedef __attribute__((ext_vector_type(8)))  float  v8f;
typedef __attribute__((ext_vector_type(4)))  unsigned int u32x4;
typedef __attribute__((ext_vector_type(8)))  int i32x8;
typedef __attribute__((ext_vector_type(4)))  int i32x4;

#define GTM 128
#define GTN 128
#define GTK 32
#define LST 40   // LDS row stride in bf16 elems (80B: 64B tile row + 16B pad)

union Frag16 { v16bf v; v8bf h[2]; };

#if __has_builtin(__builtin_amdgcn_tensor_load_to_lds)
#define HAVE_TDM 1

// D# group0: count=1 descriptor, LDS byte addr, 57b global addr, type=2.
static __device__ inline u32x4 tdm_g0(unsigned lds_addr, unsigned long long ga) {
  u32x4 g;
  g[0] = 1u;
  g[1] = lds_addr;
  g[2] = (unsigned)ga;
  g[3] = (unsigned)((ga >> 32) & 0x1FFFFFFu) | 0x80000000u;
  return g;
}
// D# group1: bf16 (data_size=1), pad_enable, pad after 16 dwords (interval=3)
// of 4 dwords (amount=3) -> LDS row stride 80B. 2D tile tile0 x tile1.
static __device__ inline i32x8 tdm_g1(unsigned td0, unsigned td1,
                                      unsigned tile0, unsigned tile1,
                                      unsigned stride0) {
  i32x8 g;
  g[0] = (int)((1u << 16) | (1u << 20) | (3u << 22) | (3u << 25));
  g[1] = (int)((td0 & 0xFFFFu) << 16);
  g[2] = (int)((td0 >> 16) | ((td1 & 0xFFFFu) << 16));
  g[3] = (int)((td1 >> 16) | (tile0 << 16));
  g[4] = (int)(tile1 & 0xFFFFu);
  g[5] = (int)stride0;
  g[6] = 0;
  g[7] = 0;
  return g;
}
static __device__ inline void tdm_load(u32x4 g0, i32x8 g1) {
  i32x4 z4 = {0, 0, 0, 0};
#if __clang_major__ >= 23
  i32x8 z8 = {0, 0, 0, 0, 0, 0, 0, 0};
  __builtin_amdgcn_tensor_load_to_lds(g0, g1, z4, z4, z8, 0);
#else
  __builtin_amdgcn_tensor_load_to_lds(g0, g1, z4, z4, 0);
#endif
}
#endif  // HAVE_TDM

// C[m,n] = sum_k Ap[m,k]*Bp[n,k]; Ap: [batch][Mp][Kp] bf16 row-major,
// Bp: [Np][Kp] bf16 (B transposed). Mp/Np mult of 128, Kp mult of 32,
// zero-padded, so the hot loop has no bounds checks. Epilogue: optional
// per-col scale/shift (fused BN or bias), residual, relu on fp32 C.
__global__ __launch_bounds__(256) void k_gemm_bf16p(
    const __bf16* __restrict__ Ap, const __bf16* __restrict__ Bp,
    long aBatch,
    float* __restrict__ C, long sCm, long sCn, long sCb,
    const float* __restrict__ Rv, long sRb,
    const float* __restrict__ scale, const float* __restrict__ shift,
    int relu, int M, int N, int Kp)
{
  __shared__ __bf16 lds[(GTM + GTN) * LST];
  __bf16* As = lds;              // As[m][k], k contiguous, stride LST
  __bf16* Bs = lds + GTM * LST;  // Bs[n][k]
  const int tid  = threadIdx.x;
  const int lane = tid & 31;
  const int wave = tid >> 5;
  const int bm = blockIdx.y * GTM, bn = blockIdx.x * GTN;
  const int wm = (wave & 3) * 32;   // 4 waves tile M
  const int wn = (wave >> 2) * 64;  // 2 waves tile N
  const int half = lane >> 4;
  const int mr   = lane & 15;

  const __bf16* Abase = Ap + (long)blockIdx.z * aBatch + (long)bm * Kp;
  const __bf16* Bbase = Bp + (long)bn * Kp;

#if defined(HAVE_TDM)
  const unsigned ldsA = (unsigned)(uintptr_t)(void*)As;
  const unsigned ldsB = (unsigned)(uintptr_t)(void*)Bs;
  const i32x8 g1 = tdm_g1((unsigned)Kp, 1u << 20, GTK, GTM, (unsigned)Kp);
#endif

  v8f acc[2][4];
  for (int i = 0; i < 2; i++)
    for (int j = 0; j < 4; j++)
      for (int r = 0; r < 8; r++) acc[i][j][r] = 0.f;

  for (int kb = 0; kb < Kp; kb += GTK) {
    __syncthreads();
#if defined(HAVE_TDM)
    if (wave == 0) {  // one wave drives the Tensor Data Mover
      tdm_load(tdm_g0(ldsA, (unsigned long long)(uintptr_t)(Abase + kb)), g1);
      tdm_load(tdm_g0(ldsB, (unsigned long long)(uintptr_t)(Bbase + kb)), g1);
      __builtin_amdgcn_s_wait_tensorcnt(0);
    }
#else
    {  // fallback: fully vectorized b128 copies (operands already bf16)
      const int row = tid >> 1, seg = (tid & 1) * 16;
      const __bf16* ga = Abase + (long)row * Kp + kb + seg;
      *(v8bf*)&As[row * LST + seg]     = *(const v8bf*)ga;
      *(v8bf*)&As[row * LST + seg + 8] = *(const v8bf*)(ga + 8);
      const __bf16* gb = Bbase + (long)row * Kp + kb + seg;
      *(v8bf*)&Bs[row * LST + seg]     = *(const v8bf*)gb;
      *(v8bf*)&Bs[row * LST + seg + 8] = *(const v8bf*)(gb + 8);
      if (kb + GTK < Kp) {
        __builtin_prefetch(ga + GTK, 0, 0);
        __builtin_prefetch(gb + GTK, 0, 0);
      }
    }
#endif
    __syncthreads();

    // A fragment: row = wm+i*16+mr; two 16B chunks K[half*8..], K[half*8+16..]
    Frag16 af[2];
    #pragma unroll
    for (int i = 0; i < 2; i++) {
      const __bf16* pr = &As[(wm + i * 16 + mr) * LST + half * 8];
      af[i].h[0] = *(const v8bf*)pr;
      af[i].h[1] = *(const v8bf*)(pr + 16);
    }
    // B fragment: col = wn+j*16+mr; two 16B chunks K[half*8..], K[half*8+16..]
    Frag16 bfm[4];
    #pragma unroll
    for (int j = 0; j < 4; j++) {
      const __bf16* pc = &Bs[(wn + j * 16 + mr) * LST + half * 8];
      bfm[j].h[0] = *(const v8bf*)pc;
      bfm[j].h[1] = *(const v8bf*)(pc + 16);
    }
    #pragma unroll
    for (int i = 0; i < 2; i++)
      #pragma unroll
      for (int j = 0; j < 4; j++)
        acc[i][j] = __builtin_amdgcn_wmma_f32_16x16x32_bf16(
            false, af[i].v, false, bfm[j].v, (short)0, acc[i][j], false, false);
  }

  // Epilogue. C/D layout: VGPR r -> row r + 8*half, col = lane&15.
  const long cb = (long)blockIdx.z * sCb;
  const long rb = (long)blockIdx.z * sRb;
  for (int j = 0; j < 4; j++) {
    const int col = bn + wn + j * 16 + mr;
    if (col >= N) continue;
    const float sc = scale ? scale[col] : 1.f;
    const float sh = shift ? shift[col] : 0.f;
    for (int i = 0; i < 2; i++) {
      const int row0 = bm + wm + i * 16 + half * 8;
      long cidx = cb + (long)row0 * sCm + (long)col * sCn;
      long ridx = rb + (long)row0 * sCm + (long)col * sCn;
      #pragma unroll
      for (int r = 0; r < 8; r++) {
        if (row0 + r < M) {
          float v = acc[i][j][r] * sc + sh;
          if (Rv)   v += Rv[ridx];
          if (relu) v = fmaxf(v, 0.f);
          C[cidx] = v;
        }
        cidx += sCm;
        ridx += sCm;
      }
    }
  }
}

// ---------------------------------------------------------------------------
// Pack kernels: strided fp32 -> padded bf16 (zero-filled pad)
// ---------------------------------------------------------------------------

// out[z][m][k] = A[z*sb + m*sm + k*sk], [Mp x Kp] per batch z
__global__ void k_packA(const float* __restrict__ A, long sm, long sk, long sb,
                        __bf16* __restrict__ out, int M, int K, int Mp, int Kp) {
  long tot = (long)Mp * Kp;
  long i = (long)blockIdx.x * blockDim.x + threadIdx.x;
  if (i >= tot) return;
  int k = i % Kp; long m = i / Kp;
  float v = 0.f;
  if (m < M && k < K) v = A[(long)blockIdx.y * sb + m * sm + (long)k * sk];
  out[(long)blockIdx.y * tot + i] = (__bf16)v;
}

// out[n][k] = B[k*sk + n*sn]  (transpose during pack), [Np x Kp]
__global__ void k_packB(const float* __restrict__ B, long sk, long sn,
                        __bf16* __restrict__ out, int N, int K, int Np, int Kp) {
  long tot = (long)Np * Kp;
  long i = (long)blockIdx.x * blockDim.x + threadIdx.x;
  if (i >= tot) return;
  int k = i % Kp; long n = i / Kp;
  float v = 0.f;
  if (n < N && k < K) v = B[(long)k * sk + n * sn];
  out[i] = (__bf16)v;
}

// im2col (KH=3 or 1) straight into packed bf16 A layout, per image z=blockIdx.y
__global__ void k_im2col_bf16(const float* __restrict__ x, long sxb,
                              __bf16* __restrict__ col, int C, int H, int W,
                              int Ho, int Wo, int KH, int stride, int pad,
                              int Kp, int Mp) {
  long tot = (long)Mp * Kp;
  long i = (long)blockIdx.x * blockDim.x + threadIdx.x;
  if (i >= tot) return;
  int KK = KH * KH, K = C * KK;
  long M = (long)Ho * Wo;
  int k = i % Kp; long m = i / Kp;
  float v = 0.f;
  if (m < M && k < K) {
    int ox = m % Wo, oy = m / Wo;
    int ci = k / KK, rr = (k % KK) / KH, cc = k % KH;
    int iy = oy * stride - pad + rr, ix = ox * stride - pad + cc;
    if (iy >= 0 && iy < H && ix >= 0 && ix < W)
      v = x[(long)blockIdx.y * sxb + ((long)ci * H + iy) * W + ix];
  }
  col[(long)blockIdx.y * tot + i] = (__bf16)v;
}

// ---------------------------------------------------------------------------
// Scalar helper kernels (bandwidth-trivial on 23.3 TB/s HBM)
// ---------------------------------------------------------------------------

__global__ void k_fill(float* p, float v, long n) {
  long i = (long)blockIdx.x * blockDim.x + threadIdx.x;
  if (i < n) p[i] = v;
}

__global__ void k_bn_coeff(const float* g, const float* b, const float* m,
                           const float* v, float* s, float* t, int n) {
  int i = blockIdx.x * blockDim.x + threadIdx.x;
  if (i < n) { float sc = g[i] * rsqrtf(v[i] + 1e-5f); s[i] = sc; t[i] = b[i] - m[i] * sc; }
}

__global__ void k_premix(const float* dm, const float* cw, const float* cb,
                         float* out, int BHW, int HW) {
  int i = blockIdx.x * blockDim.x + threadIdx.x;
  if (i >= BHW) return;
  int b = i / HW, hw = i % HW;
  const float* p = dm + (long)i * 8;
  for (int co = 0; co < 3; co++) {
    float s = cb[co];
    for (int ci = 0; ci < 8; ci++) s += cw[co * 8 + ci] * p[ci];
    out[((long)b * 3 + co) * HW + hw] = s;
  }
}

__global__ void k_conv7x7(const float* x, const float* w, const float* s,
                          const float* t, float* y, int Bn) {
  int total = Bn * 64 * 112 * 112;
  int i = blockIdx.x * blockDim.x + threadIdx.x;
  if (i >= total) return;
  int ox = i % 112, oy = (i / 112) % 112, co = (i / (112 * 112)) % 64, b = i / (64 * 112 * 112);
  float acc = 0.f;
  for (int ci = 0; ci < 3; ci++)
    for (int r = 0; r < 7; r++) {
      int iy = oy * 2 - 3 + r; if (iy < 0 || iy >= 224) continue;
      for (int c = 0; c < 7; c++) {
        int ix = ox * 2 - 3 + c; if (ix < 0 || ix >= 224) continue;
        acc += x[(((long)b * 3 + ci) * 224 + iy) * 224 + ix] * w[((co * 3 + ci) * 7 + r) * 7 + c];
      }
    }
  float v = acc * s[co] + t[co];
  y[i] = fmaxf(v, 0.f);
}

__global__ void k_maxpool(const float* x, float* y, int B, int C, int H, int W,
                          int Ho, int Wo) {
  long total = (long)B * C * Ho * Wo;
  long i = (long)blockIdx.x * blockDim.x + threadIdx.x;
  if (i >= total) return;
  int ox = i % Wo, oy = (i / Wo) % Ho;
  long bc = i / ((long)Ho * Wo);
  float m = -3.4e38f;
  for (int r = 0; r < 3; r++) {
    int iy = oy * 2 - 1 + r; if (iy < 0 || iy >= H) continue;
    for (int c = 0; c < 3; c++) {
      int ix = ox * 2 - 1 + c; if (ix < 0 || ix >= W) continue;
      m = fmaxf(m, x[(bc * H + iy) * W + ix]);
    }
  }
  y[i] = m;
}

__global__ void k_avgpool(const float* x, float* y, int BC, int S) {
  int i = blockIdx.x * blockDim.x + threadIdx.x;
  if (i >= BC) return;
  const float* p = x + (long)i * S;
  float s = 0.f;
  for (int k = 0; k < S; k++) s += p[k];
  y[i] = s / (float)S;
}

__global__ void k_rowdot(const float* x, const float* w, float* y, int R, int D) {
  int r = blockIdx.x * blockDim.x + threadIdx.x;
  if (r >= R) return;
  const float* p = x + (long)r * D;
  float s = 0.f;
  for (int d = 0; d < D; d++) s += p[d] * w[d];
  y[r] = s;
}

__device__ inline void atomicMaxFloat(float* addr, float val) {
  int* ai = (int*)addr;
  int old = *ai;
  while (true) {
    float of = __int_as_float(old);
    if (of >= val) break;
    int assumed = old;
    old = atomicCAS(ai, assumed, __float_as_int(val));
    if (old == assumed) break;
  }
}

__global__ void k_edge1(const int* src, const int* dst, const float* ss,
                        const float* sd, float* e, float* m, int E) {
  int i = blockIdx.x * blockDim.x + threadIdx.x;
  if (i >= E) return;
  float v = ss[src[i]] + sd[dst[i]];
  v = (v > 0.f) ? v : 0.2f * v;
  e[i] = v;
  atomicMaxFloat(&m[dst[i]], v);
}

__global__ void k_edge2(const int* dst, const float* e, const float* m,
                        float* den, float* ex, int E) {
  int i = blockIdx.x * blockDim.x + threadIdx.x;
  if (i >= E) return;
  float x = __expf(e[i] - m[dst[i]]);
  ex[i] = x;
  atomicAdd(&den[dst[i]], x);
}

__global__ void k_edge_scatter(const int* src, const int* dst, const float* ex,
                               const float* den, const float* hs, float* out,
                               int E, int D) {
  int e = blockIdx.x;
  if (e >= E) return;
  float a = ex[e] / (den[dst[e]] + 1e-16f);
  const float* h = hs + (long)src[e] * D;
  float* o = out + (long)dst[e] * D;
  for (int d = threadIdx.x; d < D; d += blockDim.x) atomicAdd(&o[d], a * h[d]);
}

__global__ void k_addbias(float* p, const float* b, long n, int D) {
  long i = (long)blockIdx.x * blockDim.x + threadIdx.x;
  if (i < n) p[i] += b[i % D];
}

__global__ void k_scale_relu(float* p, float sc, int relu, long n) {
  long i = (long)blockIdx.x * blockDim.x + threadIdx.x;
  if (i < n) { float v = p[i] * sc; if (relu) v = fmaxf(v, 0.f); p[i] = v; }
}

__global__ void k_concat(const float* xs, const float* img, const int* batch,
                         float* out, int R, int D1, int D2) {
  long total = (long)R * (D1 + D2);
  long i = (long)blockIdx.x * blockDim.x + threadIdx.x;
  if (i >= total) return;
  int d = i % (D1 + D2);
  long r = i / (D1 + D2);
  out[i] = (d < D1) ? xs[r * D1 + d] : img[(long)batch[r] * D2 + (d - D1)];
}

// ---------------------------------------------------------------------------
// Host orchestration
// ---------------------------------------------------------------------------

#define LAUNCH1D(kern, n, ...) do { long _n = (long)(n); \
  kern<<<(unsigned)((_n + 255) / 256), 256, 0, stream>>>(__VA_ARGS__); } while (0)

extern "C" void kernel_launch(void* const* d_in, const int* in_sizes, int n_in,
                              void* d_out, int out_size, void* d_ws, size_t ws_size,
                              hipStream_t stream) {
  (void)in_sizes; (void)n_in; (void)out_size; (void)ws_size;
  const int NN = 100000, NE = 50000, NS = 10000, Bimg = 4;

  const float* data_map = (const float*)d_in[0];
  const float* xn_in = (const float*)d_in[1];
  const float* xe_in = (const float*)d_in[2];
  const float* xs_in = (const float*)d_in[3];
  const int* e_n2n = (const int*)d_in[4];
  const int* e_n2e = (const int*)d_in[5];
  const int* e_e2n = (const int*)d_in[6];
  const int* e_s2n = (const int*)d_in[7];
  const int* e_n2s = (const int*)d_in[8];
  const int* batch_node = (const int*)d_in[9];
  const int* batch_edge = (const int*)d_in[10];
  const int* batch_source = (const int*)d_in[11];

  // ---- params in JAX pytree order (dict keys sorted lexicographically)
  int pi = 12;
  auto P = [&]() -> const float* { return (const float*)d_in[pi++]; };
  const float* cb_ = P();
  const float* cw_ = P();
  const float* fcW = P();
  const float* fcb = P();
  struct Rel { const float *Wd, *Ws, *ad, *as_, *b; };
  Rel gatp[6][5];
  for (int L = 0; L < 5; L++)
    for (int r = 0; r < 5; r++) {
      gatp[L][r].Wd = P(); gatp[L][r].Ws = P();
      gatp[L][r].ad = P(); gatp[L][r].as_ = P(); gatp[L][r].b = P();
    }
  for (int r = 0; r < 5; r++) {
    gatp[5][r].Wd = P(); gatp[5][r].Ws = P();
    gatp[5][r].ad = P(); gatp[5][r].as_ = P(); gatp[5][r].b = P();
  }
  const float* l1W = P(); const float* l1b = P();
  const float* l2W = P(); const float* l2b = P();
  struct BNp { const float *b, *g, *m, *v; };
  auto PBN = [&]() { BNp x; x.b = P(); x.g = P(); x.m = P(); x.v = P(); return x; };
  BNp bn1 = PBN();
  const float* conv1 = P();
  struct Blk { BNp b1, b2, b3, bd; const float *c1, *c2, *c3, *cd; int has_d; };
  Blk blocks[16]; int nb = 0;
  const int counts[4] = {3, 4, 6, 3};
  for (int s = 0; s < 4; s++)
    for (int i = 0; i < counts[s]; i++) {
      Blk bk{}; bk.has_d = (i == 0);
      bk.b1 = PBN(); bk.b2 = PBN(); bk.b3 = PBN();
      if (bk.has_d) bk.bd = PBN();
      bk.c1 = P(); bk.c2 = P(); bk.c3 = P();
      if (bk.has_d) bk.cd = P();
      blocks[nb++] = bk;
    }

  // ---- workspace bump allocator
  uintptr_t cur = (uintptr_t)d_ws;
  auto alloc = [&](size_t nfloats) -> float* {
    uintptr_t p = (cur + 255) & ~(uintptr_t)255;
    cur = p + nfloats * sizeof(float);
    return (float*)p;
  };
  struct BNc { float* s; float* t; };
  auto bnc = [&](BNp p, int c) -> BNc {
    BNc r; r.s = alloc(c); r.t = alloc(c);
    LAUNCH1D(k_bn_coeff, c, p.g, p.b, p.m, p.v, r.s, r.t, c);
    return r;
  };

  // packed bf16 operand arenas
  __bf16* Apack = (__bf16*)alloc((size_t)32 * 1024 * 1024);  // 64M bf16
  __bf16* Bpack = (__bf16*)alloc((size_t)8 * 1024 * 1024);   // 16M bf16

  // gemm over strided fp32 sources: pack both operands, then WMMA GEMM.
  auto gemm = [&](const float* A, long sAm, long sAk, long sAb,
                  const float* B, long sBk, long sBn,
                  float* C, long sCm, long sCn, long sCb,
                  const float* R, long sRb,
                  const float* scale, const float* shift, int relu,
                  int M, int N, int K, int batch) {
    int Mp = (M + 127) & ~127, Kp = (K + 31) & ~31, Np = (N + 127) & ~127;
    dim3 gpa((unsigned)(((long)Mp * Kp + 255) / 256), batch);
    k_packA<<<gpa, 256, 0, stream>>>(A, sAm, sAk, sAb, Apack, M, K, Mp, Kp);
    dim3 gpb((unsigned)(((long)Np * Kp + 255) / 256), 1);
    k_packB<<<gpb, 256, 0, stream>>>(B, sBk, sBn, Bpack, N, K, Np, Kp);
    dim3 g(Np / 128, Mp / 128, batch);
    k_gemm_bf16p<<<g, 256, 0, stream>>>(Apack, Bpack, (long)Mp * Kp,
                                        C, sCm, sCn, sCb, R, sRb,
                                        scale, shift, relu, M, N, Kp);
  };

  const long ACTE = (long)4 * 64 * 112 * 112;
  float* x0   = alloc((long)4 * 3 * 224 * 224);
  float* xa   = alloc(ACTE);
  float* xb   = alloc(ACTE);
  float* bufA = alloc(ACTE);
  float* bufB = alloc(ACTE);
  float* bufC = alloc(ACTE);
  float* ping = alloc(ACTE);
  float* pong = alloc(ACTE);

  // ---- CNN trunk
  LAUNCH1D(k_premix, 4 * 224 * 224, data_map, cw_, cb_, x0, 4 * 224 * 224, 224 * 224);
  BNc c1bn = bnc(bn1, 64);
  LAUNCH1D(k_conv7x7, 4 * 64 * 112 * 112, x0, conv1, c1bn.s, c1bn.t, xa, Bimg);
  LAUNCH1D(k_maxpool, (long)4 * 64 * 56 * 56, xa, xb, 4, 64, 112, 112, 56, 56);

  const float* x = xb;
  int Ci = 64, H = 56, W = 56;
  const int stage_stride[4] = {1, 2, 2, 2};
  const int widths[4] = {64, 128, 256, 512};
  int bi = 0;
  for (int s = 0; s < 4; s++) {
    for (int i = 0; i < counts[s]; i++) {
      Blk& bk = blocks[bi++];
      int w_ = widths[s], Co = 4 * w_;
      int st = (i == 0) ? stage_stride[s] : 1;
      int Ho = H / st, Wo = W / st;
      long HW = (long)H * W, HWo = (long)Ho * Wo;
      BNc B1 = bnc(bk.b1, w_), B2 = bnc(bk.b2, w_), B3 = bnc(bk.b3, Co);
      BNc Bd{nullptr, nullptr};
      if (bk.has_d) Bd = bnc(bk.bd, Co);

      // c1 (1x1 s1) z-batched, fused bn+relu
      gemm(x, 1, HW, (long)Ci * HW, bk.c1, 1, Ci,
           bufA, 1, HW, (long)w_ * HW, nullptr, 0, B1.s, B1.t, 1,
           (int)HW, w_, Ci, Bimg);
      // c2 (3x3 s=st pad1): im2col straight into packed bf16 A, z-batched
      {
        int K2 = w_ * 9, Mp = ((int)HWo + 127) & ~127, Kp = (K2 + 31) & ~31;
        int Np = (w_ + 127) & ~127;
        dim3 gi((unsigned)(((long)Mp * Kp + 255) / 256), Bimg);
        k_im2col_bf16<<<gi, 256, 0, stream>>>(bufA, (long)w_ * HW, Apack,
                                              w_, H, W, Ho, Wo, 3, st, 1, Kp, Mp);
        dim3 gpb((unsigned)(((long)Np * Kp + 255) / 256), 1);
        k_packB<<<gpb, 256, 0, stream>>>(bk.c2, 1, (long)K2, Bpack, w_, K2, Np, Kp);
        dim3 g(Np / 128, Mp / 128, Bimg);
        k_gemm_bf16p<<<g, 256, 0, stream>>>(Apack, Bpack, (long)Mp * Kp,
                                            bufB, 1, HWo, (long)w_ * HWo,
                                            nullptr, 0, B2.s, B2.t, 1,
                                            (int)HWo, w_, Kp);
      }
      // shortcut
      const float* scp; long scb;
      if (bk.has_d) {
        int Mp = ((int)HWo + 127) & ~127, Kp = (Ci + 31) & ~31;
        int Np = (Co + 127) & ~127;
        dim3 gi((unsigned)(((long)Mp * Kp + 255) / 256), Bimg);
        k_im2col_bf16<<<gi, 256, 0, stream>>>(x, (long)Ci * HW, Apack,
                                              Ci, H, W, Ho, Wo, 1, st, 0, Kp, Mp);
        dim3 gpb((unsigned)(((long)Np * Kp + 255) / 256), 1);
        k_packB<<<gpb, 256, 0, stream>>>(bk.cd, 1, (long)Ci, Bpack, Co, Ci, Np, Kp);
        dim3 g(Np / 128, Mp / 128, Bimg);
        k_gemm_bf16p<<<g, 256, 0, stream>>>(Apack, Bpack, (long)Mp * Kp,
                                            bufC, 1, HWo, (long)Co * HWo,
                                            nullptr, 0, Bd.s, Bd.t, 0,
                                            (int)HWo, Co, Kp);
        scp = bufC; scb = (long)Co * HWo;
      } else {
        scp = x; scb = (long)Ci * HW;  // Ci==Co, HW==HWo here
      }
      // c3 (1x1 s1) z-batched, fused bn + residual + relu
      float* xout = (x == ping) ? pong : ping;
      gemm(bufB, 1, HWo, (long)w_ * HWo, bk.c3, 1, w_,
           xout, 1, HWo, (long)Co * HWo, scp, scb, B3.s, B3.t, 1,
           (int)HWo, Co, w_, Bimg);
      x = xout; Ci = Co; H = Ho; W = Wo;
    }
  }
  float* pooled = alloc((long)4 * 2048);
  LAUNCH1D(k_avgpool, 4 * 2048, x, pooled, 4 * 2048, 49);
  float* img = alloc((long)4 * 256);
  gemm(pooled, 2048, 1, 0, fcW, 256, 1, img, 256, 1, 0,
       nullptr, 0, nullptr, fcb, 0, 4, 256, 2048, 1);

  // ---- graph part
  const int TS[3] = {NN, NE, NS};
  const int cnt[3] = {3, 1, 1};
  const int* batches[3] = {batch_node, batch_edge, batch_source};
  float* xcur[3]; float* xnxt[3];
  for (int t = 0; t < 3; t++) { xcur[t] = alloc((long)TS[t] * 512); xnxt[t] = alloc((long)TS[t] * 512); }
  float* hsb = alloc((long)NN * 512);
  float* hdb = alloc((long)NN * 512);
  float* ssb = alloc(NN); float* sdb = alloc(NN);
  float* mb  = alloc(NN); float* db  = alloc(NN);
  float* eb  = alloc(200000); float* exb = alloc(200000);

  hipMemcpyAsync(xcur[0], xn_in, (size_t)NN * 64 * 4, hipMemcpyDeviceToDevice, stream);
  hipMemcpyAsync(xcur[1], xe_in, (size_t)NE * 64 * 4, hipMemcpyDeviceToDevice, stream);
  hipMemcpyAsync(xcur[2], xs_in, (size_t)NS * 64 * 4, hipMemcpyDeviceToDevice, stream);

  struct RelT { int st, dt; const int* e; int E; };
  const RelT rt[5] = {
      {0, 0, e_n2n, 200000}, {0, 1, e_n2e, 100000}, {1, 0, e_e2n, 100000},
      {2, 0, e_s2n, 50000},  {0, 2, e_n2s, 50000}};
  const int pmap[5] = {2, 1, 0, 4, 3};  // RELS order -> sorted-param-key index

  auto hetero = [&](Rel* lp, int din, int dout, int relu) {
    for (int t = 0; t < 3; t++) LAUNCH1D(k_fill, (long)TS[t] * dout, xnxt[t], 0.f, (long)TS[t] * dout);
    for (int r = 0; r < 5; r++) {
      const RelT& R0 = rt[r];
      const Rel& pr = lp[pmap[r]];
      int Ns = TS[R0.st], Nd = TS[R0.dt];
      gemm(xcur[R0.st], din, 1, 0, pr.Ws, dout, 1, hsb, dout, 1, 0,
           nullptr, 0, nullptr, nullptr, 0, Ns, dout, din, 1);
      gemm(xcur[R0.dt], din, 1, 0, pr.Wd, dout, 1, hdb, dout, 1, 0,
           nullptr, 0, nullptr, nullptr, 0, Nd, dout, din, 1);
      LAUNCH1D(k_rowdot, Ns, hsb, pr.as_, ssb, Ns, dout);
      LAUNCH1D(k_rowdot, Nd, hdb, pr.ad, sdb, Nd, dout);
      LAUNCH1D(k_fill, Nd, mb, -1e30f, (long)Nd);
      LAUNCH1D(k_fill, Nd, db, 0.f, (long)Nd);
      const int* src = R0.e;
      const int* dst = R0.e + R0.E;
      LAUNCH1D(k_edge1, R0.E, src, dst, ssb, sdb, eb, mb, R0.E);
      LAUNCH1D(k_edge2, R0.E, dst, eb, mb, db, exb, R0.E);
      k_edge_scatter<<<R0.E, 128, 0, stream>>>(src, dst, exb, db, hsb,
                                               xnxt[R0.dt], R0.E, dout);
      LAUNCH1D(k_addbias, (long)Nd * dout, xnxt[R0.dt], pr.b, (long)Nd * dout, dout);
    }
    for (int t = 0; t < 3; t++) {
      LAUNCH1D(k_scale_relu, (long)TS[t] * dout, xnxt[t], 1.f / cnt[t], relu, (long)TS[t] * dout);
      float* tmp = xcur[t]; xcur[t] = xnxt[t]; xnxt[t] = tmp;
    }
  };

  // gat_top (64->64, no relu), then concat with img[batch] -> 320
  hetero(gatp[5], 64, 64, 0);
  for (int t = 0; t < 3; t++) {
    LAUNCH1D(k_concat, (long)TS[t] * 320, xcur[t], img, batches[t], xnxt[t], TS[t], 64, 256);
    float* tmp = xcur[t]; xcur[t] = xnxt[t]; xnxt[t] = tmp;
  }
  const int din[5]  = {320, 128, 256, 512, 256};
  const int dout[5] = {128, 256, 512, 256, 128};
  for (int L = 0; L < 5; L++) hetero(gatp[L], din[L], dout[L], 1);

  // final MLP on node features -> d_out [100000, 2]
  float* h = hsb;  // reuse
  gemm(xcur[0], 128, 1, 0, l1W, 256, 1, h, 256, 1, 0,
       nullptr, 0, nullptr, l1b, 1, NN, 256, 128, 1);
  gemm(h, 256, 1, 0, l2W, 2, 1, (float*)d_out, 2, 1, 0,
       nullptr, 0, nullptr, l2b, 0, NN, 2, 256, 1);
}
